// CODABlocks_29145648071300
// MI455X (gfx1250) — matrix operand order repaired and checked by reading, not compiled, not run on hardware
//
#include <hip/hip_runtime.h>
#include <hip/hip_bf16.h>

// CODABlocks forward for MI455X (gfx1250, wave32, WMMA).
// All truncated FFT/iFFT stages are expressed as batched GEMMs against
// precomputed DFT cos/sin bases so they run on v_wmma_f32_16x16x32_f16.

typedef __attribute__((ext_vector_type(16))) _Float16 v16h;
typedef __attribute__((ext_vector_type(8)))  float    v8f;

// ---------------- problem constants ----------------
constexpr int NTOK = 128;        // b*t
constexpr int HW   = 16384;      // 128*128
constexpr int NH   = 16;
constexpr int BH   = 2048;       // NTOK*NH

// ---------------- DFT table offsets (floats, inside TAB region) ----------
constexpr int T_CW33 = 0;                       // [128w][33kw]  fwd cos /128
constexpr int T_SW33 = T_CW33 + 128*33;         // [128w][33kw]  fwd -sin /128
constexpr int T_CHF  = T_SW33 + 128*33;         // [128kh][128h] fwd cos /128
constexpr int T_SHF  = T_CHF  + 16384;          // [128kh][128h] fwd sin /128
constexpr int T_CI64 = T_SHF  + 16384;          // [64h][64kh]   inv cos
constexpr int T_SI64 = T_CI64 + 4096;           // [64h][64kh]   inv sin
constexpr int T_CB64 = T_SI64 + 4096;           // [33kw][64w]   inv a*cos
constexpr int T_SB64 = T_CB64 + 33*64;          // [33kw][64w]   inv -a*sin
constexpr int T_CIV  = T_SB64 + 33*64;          // [128h][16j]   V inv rows cos
constexpr int T_SIV  = T_CIV  + 2048;
constexpr int T_CB9  = T_SIV  + 2048;           // [9kw][128w]
constexpr int T_SB9  = T_CB9  + 9*128;
constexpr int T_CH32 = T_SB9  + 9*128;          // [32j][128h]   fwd gathered /128
constexpr int T_SH32 = T_CH32 + 4096;
constexpr int T_CI32 = T_SH32 + 4096;           // [128h][32j]   inv gathered
constexpr int T_SI32 = T_CI32 + 4096;
constexpr int T_CB17 = T_SI32 + 4096;           // [17kw][128w]
constexpr int T_SB17 = T_CB17 + 17*128;
constexpr int T_END  = T_SB17 + 17*128;         // = 80768 floats

// =====================================================================
// Generic batched GEMM on WMMA: C = alpha*A(MxK)*B(KxN) + beta*C.
// One wave per 16x16 tile, K in steps of 32, f16 fragments / f32 acc.
// strides sA/sB/sC in elements (0 => operand shared across batch).
//
// Branch-free hot loop: M/N edges handled by address clamping (garbage
// rows/cols are never stored), K tail handled once with cndmask zero-pad.
// =====================================================================
__global__ __launch_bounds__(128)
void wmma_gemm_kernel(const float* __restrict__ A, const float* __restrict__ B,
                      float* __restrict__ C, int M, int N, int K,
                      int lda, int ldb, int ldc,
                      long long sA, long long sB, long long sC,
                      float alpha, float beta, int transB)
{
    const int batch = blockIdx.y;
    const float* Ab = A + (long long)batch * sA;
    const float* Bb = B + (long long)batch * sB;
    float*       Cb = C + (long long)batch * sC;
    const int lane   = threadIdx.x & 31;
    const int wid    = threadIdx.x >> 5;
    const int tilesN = (N + 15) >> 4;
    const int tilesM = (M + 15) >> 4;
    const int tile   = blockIdx.x * 4 + wid;
    if (tile >= tilesM * tilesN) return;
    const int tm = (tile / tilesN) << 4;
    const int tn = (tile % tilesN) << 4;

    // A fragment: lanes 0-15 row M=lane, K 0..7 (V0..3) & 16..23 (V4..7);
    //             lanes 16-31 same rows, K 8..15 & 24..31.
    const int ar  = lane & 15;
    const int ak0 = (lane >> 4) << 3;
    // B fragment: lanes 0-15 col N=lane, K 0..7 & 16..23; lanes 16-31 K 8..15 & 24..31.
    const int bn  = lane & 15;
    const int bk0 = (lane >> 4) << 3;
    const int row = tm + ar;
    const int col = tn + bn;

    // Clamped addressing: out-of-range rows/cols compute garbage that is
    // simply never stored, so no predication is needed in the K loop.
    const int rowc = (row < M) ? row : (M - 1);
    const int colc = (col < N) ? col : (N - 1);
    const float* __restrict__ Ar = Ab + (long long)rowc * lda;              // A row base
    const float* __restrict__ Bc = transB ? (Bb + (long long)colc * ldb)    // B col base
                                          : (Bb + colc);
    const long long bstep = transB ? 1 : ldb;   // stride between consecutive k in B

    v8f acc = {};
    int k0 = 0;
    const int Kmain = K & ~31;
    for (; k0 < Kmain; k0 += 32) {
        // prefetch next K-slice (speculative; OOB prefetches are dropped)
        __builtin_prefetch(Ar + k0 + 32);
        __builtin_prefetch(Bc + (long long)(k0 + 32) * bstep);
        v16h af, bf;
#pragma unroll
        for (int v = 0; v < 8; ++v) {
            const int kk = (v < 4) ? (ak0 + 2*v) : (16 + ak0 + 2*(v-4));
            af[2*v]   = (_Float16)Ar[k0 + kk];
            af[2*v+1] = (_Float16)Ar[k0 + kk + 1];
        }
#pragma unroll
        for (int v = 0; v < 8; ++v) {
            const int kB = k0 + ((v < 4) ? (bk0 + 2*v) : (16 + bk0 + 2*(v-4)));
            bf[2*v]   = (_Float16)Bc[(long long)kB * bstep];
            bf[2*v+1] = (_Float16)Bc[(long long)(kB + 1) * bstep];
        }
        acc = __builtin_amdgcn_wmma_f32_16x16x32_f16(false, af, false, bf,
                                                     (short)0, acc, false, false);
    }
    if (k0 < K) {       // K tail: clamp addresses, zero-pad out-of-range k
        v16h af, bf;
#pragma unroll
        for (int v = 0; v < 8; ++v) {
            const int kk = (v < 4) ? (ak0 + 2*v) : (16 + ak0 + 2*(v-4));
            const int ka0 = k0 + kk, ka1 = k0 + kk + 1;
            const float x0 = Ar[(ka0 < K) ? ka0 : (K - 1)];
            const float x1 = Ar[(ka1 < K) ? ka1 : (K - 1)];
            af[2*v]   = (_Float16)((ka0 < K) ? x0 : 0.f);
            af[2*v+1] = (_Float16)((ka1 < K) ? x1 : 0.f);
        }
#pragma unroll
        for (int v = 0; v < 8; ++v) {
            const int kb0 = k0 + ((v < 4) ? (bk0 + 2*v) : (16 + bk0 + 2*(v-4))), kb1 = kb0 + 1;
            const float x0 = Bc[(long long)((kb0 < K) ? kb0 : (K - 1)) * bstep];
            const float x1 = Bc[(long long)((kb1 < K) ? kb1 : (K - 1)) * bstep];
            bf[2*v]   = (_Float16)((kb0 < K) ? x0 : 0.f);
            bf[2*v+1] = (_Float16)((kb1 < K) ? x1 : 0.f);
        }
        acc = __builtin_amdgcn_wmma_f32_16x16x32_f16(false, af, false, bf,
                                                     (short)0, acc, false, false);
    }

    // C/D layout: VGPR v -> M=v (lanes 0-15) / M=8+v (lanes 16-31), N=lane&15.
    if (col < N) {
        const int mb = tm + ((lane >> 4) << 3);
#pragma unroll
        for (int v = 0; v < 8; ++v) {
            const int m = mb + v;
            if (m < M) {
                const long long ci = (long long)m*ldc + col;
                const float r = alpha * acc[v];
                Cb[ci] = (beta == 0.f) ? r : fmaf(beta, Cb[ci], r);
            }
        }
    }
}

// =====================================================================
// DFT basis tables
// =====================================================================
__global__ void build_tables_kernel(float* __restrict__ T)
{
    const int i = blockIdx.x * blockDim.x + threadIdx.x;
    const float PI2 = 6.28318530717958647692f;
    if (i < 128*33) {                       // fwd along W (cols 0..32), /128
        int w = i / 33, kw = i - w*33;
        float th = PI2 * (float)(w*kw) / 128.f;
        T[T_CW33+i] = cosf(th) * (1.f/128.f);
        T[T_SW33+i] = -sinf(th) * (1.f/128.f);
    }
    if (i < 16384) {                        // fwd along H (all rows), /128
        int kh = i >> 7, h = i & 127;
        float th = PI2 * (float)(kh*h) / 128.f;
        T[T_CHF+i] = cosf(th) * (1.f/128.f);
        T[T_SHF+i] = sinf(th) * (1.f/128.f);
    }
    if (i < 4096) {                         // inv rows at 64
        int h = i >> 6, kh = i & 63;
        float th = PI2 * (float)(h*kh) / 64.f;
        T[T_CI64+i] = cosf(th);
        T[T_SI64+i] = sinf(th);
    }
    if (i < 33*64) {                        // inv cols at 64 (hermitian weights)
        int kw = i >> 6, w = i & 63;
        float a = (kw == 0 || kw == 32) ? 1.f : 2.f;
        float th = PI2 * (float)(kw*w) / 64.f;
        T[T_CB64+i] =  a * cosf(th);
        T[T_SB64+i] = -a * sinf(th);
    }
    if (i < 2048) {                         // V inv rows, gathered kh {0..7,120..127}
        int h = i >> 4, j = i & 15;
        int kh = (j < 8) ? j : 120 + (j - 8);
        float th = PI2 * (float)(h*kh) / 128.f;
        T[T_CIV+i] = cosf(th);
        T[T_SIV+i] = sinf(th);
    }
    if (i < 9*128) {                        // inv cols at 128, 9 kept cols
        int kw = i >> 7, w = i & 127;
        float a = (kw == 0) ? 1.f : 2.f;
        float th = PI2 * (float)(kw*w) / 128.f;
        T[T_CB9+i] =  a * cosf(th);
        T[T_SB9+i] = -a * sinf(th);
    }
    if (i < 4096) {                         // fwd rows gathered {0..15,112..127}, /128
        int j = i >> 7, h = i & 127;
        int kh = (j < 16) ? j : 112 + (j - 16);
        float th = PI2 * (float)(kh*h) / 128.f;
        T[T_CH32+i] = cosf(th) * (1.f/128.f);
        T[T_SH32+i] = sinf(th) * (1.f/128.f);
    }
    if (i < 4096) {                         // inv rows gathered
        int h = i >> 5, j = i & 31;
        int kh = (j < 16) ? j : 112 + (j - 16);
        float th = PI2 * (float)(kh*h) / 128.f;
        T[T_CI32+i] = cosf(th);
        T[T_SI32+i] = sinf(th);
    }
    if (i < 17*128) {                       // inv cols at 128, 17 kept cols
        int kw = i >> 7, w = i & 127;
        float a = (kw == 0) ? 1.f : 2.f;
        float th = PI2 * (float)(kw*w) / 128.f;
        T[T_CB17+i] =  a * cosf(th);
        T[T_SB17+i] = -a * sinf(th);
    }
}

// =====================================================================
// Elementwise / reduction kernels
// =====================================================================
__global__ __launch_bounds__(256)
void instnorm_kernel(const float* __restrict__ in, const float* __restrict__ addin,
                     float* __restrict__ out, const float* g, const float* b)
{
    __shared__ float red[256];
    const long long base = (long long)blockIdx.x * HW;
    float sum = 0.f, sq = 0.f;
    for (int i = threadIdx.x; i < HW; i += 256) {
        float v = in[base+i]; if (addin) v += addin[base+i];
        sum += v; sq += v*v;
    }
    red[threadIdx.x] = sum; __syncthreads();
    for (int o = 128; o > 0; o >>= 1) { if (threadIdx.x < o) red[threadIdx.x] += red[threadIdx.x+o]; __syncthreads(); }
    const float mu = red[0] * (1.f/HW);
    __syncthreads();
    red[threadIdx.x] = sq; __syncthreads();
    for (int o = 128; o > 0; o >>= 1) { if (threadIdx.x < o) red[threadIdx.x] += red[threadIdx.x+o]; __syncthreads(); }
    const float var = red[0] * (1.f/HW) - mu*mu;
    const float inv = rsqrtf(var + 1e-5f);
    const float gg = g[0], bb = b[0];
    for (int i = threadIdx.x; i < HW; i += 256) {
        float v = in[base+i]; if (addin) v += addin[base+i];
        out[base+i] = (v - mu) * inv * gg + bb;
    }
}

__global__ void softmax32_kernel(float* __restrict__ S, int rows)
{
    const int row  = blockIdx.x * 8 + (threadIdx.x >> 5);
    const int lane = threadIdx.x & 31;
    if (row >= rows) return;
    float v = S[row*32 + lane] * (1.f/64.f);   // scale = sqrt(4096)*T
    float m = v;
    for (int k = 16; k > 0; k >>= 1) m = fmaxf(m, __shfl_xor(m, k, 32));
    float e = __expf(v - m);
    float s = e;
    for (int k = 16; k > 0; k >>= 1) s += __shfl_xor(s, k, 32);
    S[row*32 + lane] = e / s;
}

// assembled K/Q spectrum at 64x33: resampled skip (a*X) + spectral weights + DC bias
__global__ void assemble_kq_kernel(const float* __restrict__ XFre, const float* __restrict__ XFim,
                                   const float* __restrict__ w, const float* skw, const float* skb,
                                   float* __restrict__ Fre, float* __restrict__ Fim)
{
    const int i = blockIdx.x * blockDim.x + threadIdx.x;
    if (i >= BH * 64 * 33) return;
    const int u = i / (64*33); const int rem = i - u*(64*33);
    const int r = rem / 33, c = rem - r*33;
    const int tok = u >> 4, h = u & 15;
    const int kh = (r < 32) ? r : r + 64;           // map 64-grid row -> 128-grid row
    const float xr = XFre[tok*4224 + kh*33 + c];
    const float xi = XFim[tok*4224 + kh*33 + c];
    const float a = skw[h];
    float vr = a*xr, vi = a*xi;
    if (c < 9 && (r < 8 || r >= 56)) {              // spectral weight region
        const int j  = (r < 8) ? r : 8 + (r - 56);
        const int wi = ((h*16 + j)*9 + c)*2;
        const float wr = w[wi], wm = w[wi+1];
        vr += wr*xr - wm*xi;
        vi += wr*xi + wm*xr;
    }
    if (r == 0 && c == 0) vr += skb[h];
    Fre[i] = vr; Fim[i] = vi;
}

__global__ void assemble_v_kernel(const float* __restrict__ XFre, const float* __restrict__ XFim,
                                  const float* __restrict__ w,
                                  float* __restrict__ Fre, float* __restrict__ Fim)
{
    const int i = blockIdx.x * blockDim.x + threadIdx.x;
    if (i >= BH * 144) return;
    const int u = i / 144; const int rem = i - u*144;
    const int j = rem / 9, c = rem - j*9;
    const int tok = u >> 4, h = u & 15;
    const int kh = (j < 8) ? j : 120 + (j - 8);
    const float xr = XFre[tok*4224 + kh*33 + c];
    const float xi = XFim[tok*4224 + kh*33 + c];
    const int wi = ((h*16 + j)*9 + c)*2;
    const float wr = w[wi], wm = w[wi+1];
    Fre[i] = wr*xr - wm*xi;
    Fim[i] = wr*xi + wm*xr;
}

// dir 0: [tok*16+h][D] -> [(b*16+h)*32+t][D] ; dir 1: inverse
__global__ void permute_kernel(const float* __restrict__ in, float* __restrict__ out,
                               int D, int dir, long long total)
{
    const long long i = (long long)blockIdx.x * 256 + threadIdx.x;
    if (i >= total) return;
    const int u = (int)(i / D), p = (int)(i - (long long)u * D);
    int dstU;
    if (dir == 0) { int tok = u >> 4, h = u & 15, b = tok >> 5, t = tok & 31; dstU = (b*16+h)*32 + t; }
    else          { int t = u & 31, bh = u >> 5, h = bh & 15, b = bh >> 4;    dstU = (b*32+t)*16 + h; }
    out[(long long)dstU * D + p] = in[i];
}

// V = spec + a_h * tokens_norm + b_h, written directly into attention layout
__global__ void finish_v_kernel(const float* __restrict__ VSp, const float* __restrict__ TNp,
                                const float* skw, const float* skb, float* __restrict__ VAp)
{
    const long long i = (long long)blockIdx.x * 256 + threadIdx.x;
    if (i >= (long long)BH * HW) return;
    const int u = (int)(i >> 14), p = (int)(i & 16383);
    const int tok = u >> 4, h = u & 15, b = tok >> 5, t = tok & 31;
    const float v = VSp[i] + skw[h] * TNp[(long long)tok*HW + p] + skb[h];
    VAp[(long long)((b*16+h)*32 + t) * HW + p] = v;
}

// sum over heads of per-head attn spectra times proj weights
__global__ void combine_proj_kernel(const float* __restrict__ Xre, const float* __restrict__ Xim,
                                    const float* __restrict__ pw,
                                    float* __restrict__ Mre, float* __restrict__ Mim)
{
    const int i = blockIdx.x * blockDim.x + threadIdx.x;
    if (i >= NTOK * 544) return;
    const int tok = i / 544, rem = i - tok*544;
    const int j = rem / 17, c = rem - j*17;
    float ar = 0.f, ai = 0.f;
    for (int h = 0; h < 16; ++h) {
        const int xi = (tok*16 + h)*544 + rem;
        const float xr = Xre[xi], xm = Xim[xi];
        const int wi = ((h*32 + j)*17 + c)*2;
        const float wr = pw[wi], wm = pw[wi+1];
        ar += wr*xr - wm*xm;
        ai += wr*xm + wm*xr;
    }
    Mre[i] = ar; Mim[i] = ai;
}

__global__ void mixer_modes_kernel(const float* __restrict__ Xre, const float* __restrict__ Xim,
                                   const float* __restrict__ w,
                                   float* __restrict__ Mre, float* __restrict__ Mim)
{
    const int i = blockIdx.x * blockDim.x + threadIdx.x;
    if (i >= NTOK * 544) return;
    const int rem = i % 544;
    const float xr = Xre[i], xm = Xim[i];
    const float wr = w[rem*2], wm = w[rem*2 + 1];
    Mre[i] = wr*xr - wm*xm;
    Mim[i] = wr*xm + wm*xr;
}

// proj_out = spec + sum_h skw[h]*attn_h + skb + tokens_raw
__global__ void proj_skip_kernel(const float* __restrict__ PSp, const float* __restrict__ ATTPp,
                                 const float* pskw, const float* pskb,
                                 const float* __restrict__ xraw, float* __restrict__ out)
{
    const long long i = (long long)blockIdx.x * 256 + threadIdx.x;
    if (i >= (long long)NTOK * HW) return;
    const int tok = (int)(i >> 14), p = (int)(i & 16383);
    float s = PSp[i] + pskb[0] + xraw[i];
    for (int h = 0; h < 16; ++h)
        s += pskw[h] * ATTPp[(long long)(tok*16 + h)*HW + p];
    out[i] = s;
}

__global__ void skip_add_act_kernel(const float* __restrict__ xf, const float* __restrict__ xin,
                                    const float* aw, const float* ab, float* __restrict__ out,
                                    long long n, int do_gelu)
{
    const long long i = (long long)blockIdx.x * 256 + threadIdx.x;
    if (i >= n) return;
    float y = xf[i] + aw[0]*xin[i] + ab[0];
    if (do_gelu) y = 0.5f * y * (1.f + erff(y * 0.70710678118654752f));
    out[i] = y;
}

__global__ void add_kernel(const float* __restrict__ a, const float* __restrict__ b,
                           float* __restrict__ out, long long n)
{
    const long long i = (long long)blockIdx.x * 256 + threadIdx.x;
    if (i >= n) return;
    out[i] = a[i] + b[i];
}

// =====================================================================
// Host side
// =====================================================================
static inline unsigned gblocks(long long n) { return (unsigned)((n + 255) / 256); }

static void gemm(hipStream_t st, const float* A, const float* B, float* C,
                 int M, int N, int K, int lda, int ldb, int ldc,
                 long long sA, long long sB, long long sC, int batch,
                 float alpha, float beta, int transB)
{
    const int tiles = ((M + 15) / 16) * ((N + 15) / 16);
    dim3 grid((unsigned)((tiles + 3) / 4), (unsigned)batch, 1);
    hipLaunchKernelGGL(wmma_gemm_kernel, grid, dim3(128), 0, st,
                       A, B, C, M, N, K, lda, ldb, ldc, sA, sB, sC, alpha, beta, transB);
}

extern "C" void kernel_launch(void* const* d_in, const int* in_sizes, int n_in,
                              void* d_out, int out_size, void* d_ws, size_t ws_size,
                              hipStream_t stream)
{
    (void)in_sizes; (void)n_in; (void)out_size; (void)ws_size;
    const float* x        = (const float*)d_in[0];
    const float* key_w    = (const float*)d_in[1];
    const float* key_skw  = (const float*)d_in[2];
    const float* key_skb  = (const float*)d_in[3];
    const float* qry_w    = (const float*)d_in[4];
    const float* qry_skw  = (const float*)d_in[5];
    const float* qry_skb  = (const float*)d_in[6];
    const float* val_w    = (const float*)d_in[7];
    const float* val_skw  = (const float*)d_in[8];
    const float* val_skb  = (const float*)d_in[9];
    const float* proj_w   = (const float*)d_in[10];
    const float* proj_skw = (const float*)d_in[11];
    const float* proj_skb = (const float*)d_in[12];
    const float* n1g = (const float*)d_in[13], *n1b = (const float*)d_in[14];
    const float* ang = (const float*)d_in[15], *anb = (const float*)d_in[16];
    const float* n2g = (const float*)d_in[17], *n2b = (const float*)d_in[18];
    const float* mw1 = (const float*)d_in[19], *msw1 = (const float*)d_in[20], *msb1 = (const float*)d_in[21];
    const float* mng1= (const float*)d_in[22], *mnb1 = (const float*)d_in[23];
    const float* mw2 = (const float*)d_in[24], *msw2 = (const float*)d_in[25], *msb2 = (const float*)d_in[26];
    const float* mng2= (const float*)d_in[27], *mnb2 = (const float*)d_in[28];
    const float* mog = (const float*)d_in[29], *mob  = (const float*)d_in[30];
    float* outp = (float*)d_out;

    // ---------------- workspace layout ----------------
    float* Wk = (float*)d_ws;
    size_t off = 0;
    auto take = [&](size_t n) { float* p = Wk + off; off += n; return p; };
    float* TAB  = take(T_END);
    float* TN   = take((size_t)NTOK * HW);            // tokens_norm
    float* XFre = take((size_t)NTOK * 128 * 33);      // tokens spectrum
    float* XFim = take((size_t)NTOK * 128 * 33);
    float* SCRA = take((size_t)BH * 2176 * 2);        // stage-A / row-iDFT scratch
    float* MK   = take((size_t)BH * 2112 * 2);        // assembled modes
    float* G    = take((size_t)BH * 2112 * 2);        // row-stage outputs / attn spectra
    float* KS   = take((size_t)BH * 4096);            // spatial K/Q scratch
    float* KA   = take((size_t)BH * 4096);
    float* QA   = take((size_t)BH * 4096);
    float* VA   = take((size_t)BH * HW);              // V (attention layout) / ATTP
    float* VS   = take((size_t)BH * HW);              // V spec scratch / ATT
    float* SMx  = take((size_t)64 * 1024);            // attention scores
    float* PS   = take((size_t)NTOK * HW);            // spectral branch output
    float* U0   = take((size_t)NTOK * HW);
    float* U1   = take((size_t)NTOK * HW);             // attn residual output
    float* U2   = take((size_t)NTOK * HW);             // mixer input m0
    float* U3   = take((size_t)NTOK * HW);             // normed spec tmp
    float* U4   = take((size_t)NTOK * HW);             // mixer1 output
    float* U5   = take((size_t)NTOK * HW);             // mixer2 output

    const long long NHW = (long long)NTOK * HW;

    // 0) DFT tables
    hipLaunchKernelGGL(build_tables_kernel, dim3(64), dim3(256), 0, stream, TAB);

    // 1) tokens_norm = IN(x)
    hipLaunchKernelGGL(instnorm_kernel, dim3(NTOK), dim3(256), 0, stream,
                       x, (const float*)nullptr, TN, n1g, n1b);

    // 2) truncated forward spectrum of tokens_norm: 128 rows x 33 cols
    float* YAre = SCRA, *YAim = SCRA + (size_t)NTOK * 4224;
    gemm(stream, TN, TAB+T_CW33, YAre, 128,33,128, 128,33,33, HW,0,4224, NTOK, 1.f,0.f,0);
    gemm(stream, TN, TAB+T_SW33, YAim, 128,33,128, 128,33,33, HW,0,4224, NTOK, 1.f,0.f,0);
    gemm(stream, TAB+T_CHF, YAre, XFre, 128,33,128, 128,33,33, 0,4224,4224, NTOK,  1.f,0.f,0);
    gemm(stream, TAB+T_SHF, YAim, XFre, 128,33,128, 128,33,33, 0,4224,4224, NTOK,  1.f,1.f,0);
    gemm(stream, TAB+T_CHF, YAim, XFim, 128,33,128, 128,33,33, 0,4224,4224, NTOK,  1.f,0.f,0);
    gemm(stream, TAB+T_SHF, YAre, XFim, 128,33,128, 128,33,33, 0,4224,4224, NTOK, -1.f,1.f,0);

    // 3) K and Q paths: assemble 64x33 spectrum -> iDFT(64x64) -> permute
    auto kq_path = [&](const float* w, const float* skw, const float* skb, float* dstA) {
        float* Fre = MK, *Fim = MK + (size_t)BH * 2112;
        hipLaunchKernelGGL(assemble_kq_kernel, dim3(gblocks((long long)BH*2112)), dim3(256), 0, stream,
                           XFre, XFim, w, skw, skb, Fre, Fim);
        float* Gre = G, *Gim = G + (size_t)BH * 2112;
        gemm(stream, TAB+T_CI64, Fre, Gre, 64,33,64, 64,33,33, 0,2112,2112, BH,  1.f,0.f,0);
        gemm(stream, TAB+T_SI64, Fim, Gre, 64,33,64, 64,33,33, 0,2112,2112, BH, -1.f,1.f,0);
        gemm(stream, TAB+T_SI64, Fre, Gim, 64,33,64, 64,33,33, 0,2112,2112, BH,  1.f,0.f,0);
        gemm(stream, TAB+T_CI64, Fim, Gim, 64,33,64, 64,33,33, 0,2112,2112, BH,  1.f,1.f,0);
        gemm(stream, Gre, TAB+T_CB64, KS, 64,64,33, 33,64,64, 2112,0,4096, BH, 1.f,0.f,0);
        gemm(stream, Gim, TAB+T_SB64, KS, 64,64,33, 33,64,64, 2112,0,4096, BH, 1.f,1.f,0);
        hipLaunchKernelGGL(permute_kernel, dim3(gblocks((long long)BH*4096)), dim3(256), 0, stream,
                           KS, dstA, 4096, 0, (long long)BH*4096);
    };
    kq_path(key_w, key_skw, key_skb, KA);
    kq_path(qry_w, qry_skw, qry_skb, QA);

    // 4) V path: 16x9 modes -> iDFT(128x128) + spatial skip, attention layout
    {
        float* Fre = MK, *Fim = MK + (size_t)BH * 144;
        hipLaunchKernelGGL(assemble_v_kernel, dim3(gblocks((long long)BH*144)), dim3(256), 0, stream,
                           XFre, XFim, val_w, Fre, Fim);
        float* Gre = G, *Gim = G + (size_t)BH * 1152;
        gemm(stream, TAB+T_CIV, Fre, Gre, 128,9,16, 16,9,9, 0,144,1152, BH,  1.f,0.f,0);
        gemm(stream, TAB+T_SIV, Fim, Gre, 128,9,16, 16,9,9, 0,144,1152, BH, -1.f,1.f,0);
        gemm(stream, TAB+T_SIV, Fre, Gim, 128,9,16, 16,9,9, 0,144,1152, BH,  1.f,0.f,0);
        gemm(stream, TAB+T_CIV, Fim, Gim, 128,9,16, 16,9,9, 0,144,1152, BH,  1.f,1.f,0);
        gemm(stream, Gre, TAB+T_CB9, VS, 128,128,9, 9,128,128, 1152,0,HW, BH, 1.f,0.f,0);
        gemm(stream, Gim, TAB+T_SB9, VS, 128,128,9, 9,128,128, 1152,0,HW, BH, 1.f,1.f,0);
        hipLaunchKernelGGL(finish_v_kernel, dim3(gblocks((long long)BH*HW)), dim3(256), 0, stream,
                           VS, TN, val_skw, val_skb, VA);
    }

    // 5) attention: S = Q K^T (/64 in softmax), P = softmax(S), ATT = P V
    gemm(stream, QA, KA, SMx, 32,32,4096, 4096,4096,32, 131072,131072,1024, 64, 1.f,0.f,1);
    hipLaunchKernelGGL(softmax32_kernel, dim3(256), dim3(256), 0, stream, SMx, 2048);
    float* ATT = VS;                     // reuse V spec scratch
    gemm(stream, SMx, VA, ATT, 32,16384,32, 32,16384,16384, 1024,524288,524288, 64, 1.f,0.f,0);
    float* ATTP = VA;                    // reuse V buffer, [tok*16+h][HW]
    hipLaunchKernelGGL(permute_kernel, dim3(gblocks((long long)BH*HW)), dim3(256), 0, stream,
                       ATT, ATTP, HW, 1, (long long)BH*HW);

    // 6) projection FNO layer (16 -> 1), modes 32x17
    {
        float* Y2re = SCRA, *Y2im = SCRA + (size_t)BH * 2176;
        gemm(stream, ATTP, TAB+T_CW33, Y2re, 128,17,128, 128,33,17, HW,0,2176, BH, 1.f,0.f,0);
        gemm(stream, ATTP, TAB+T_SW33, Y2im, 128,17,128, 128,33,17, HW,0,2176, BH, 1.f,0.f,0);
        float* XAre = G, *XAim = G + (size_t)BH * 544;
        gemm(stream, TAB+T_CH32, Y2re, XAre, 32,17,128, 128,17,17, 0,2176,544, BH,  1.f,0.f,0);
        gemm(stream, TAB+T_SH32, Y2im, XAre, 32,17,128, 128,17,17, 0,2176,544, BH,  1.f,1.f,0);
        gemm(stream, TAB+T_CH32, Y2im, XAim, 32,17,128, 128,17,17, 0,2176,544, BH,  1.f,0.f,0);
        gemm(stream, TAB+T_SH32, Y2re, XAim, 32,17,128, 128,17,17, 0,2176,544, BH, -1.f,1.f,0);
        float* Mre = MK, *Mim = MK + (size_t)NTOK * 544;
        hipLaunchKernelGGL(combine_proj_kernel, dim3(gblocks((long long)NTOK*544)), dim3(256), 0, stream,
                           XAre, XAim, proj_w, Mre, Mim);
        float* Gre = SCRA, *Gim = SCRA + (size_t)NTOK * 2176;
        gemm(stream, TAB+T_CI32, Mre, Gre, 128,17,32, 32,17,17, 0,544,2176, NTOK,  1.f,0.f,0);
        gemm(stream, TAB+T_SI32, Mim, Gre, 128,17,32, 32,17,17, 0,544,2176, NTOK, -1.f,1.f,0);
        gemm(stream, TAB+T_SI32, Mre, Gim, 128,17,32, 32,17,17, 0,544,2176, NTOK,  1.f,0.f,0);
        gemm(stream, TAB+T_CI32, Mim, Gim, 128,17,32, 32,17,17, 0,544,2176, NTOK,  1.f,1.f,0);
        gemm(stream, Gre, TAB+T_CB17, PS, 128,128,17, 17,128,128, 2176,0,HW, NTOK, 1.f,0.f,0);
        gemm(stream, Gim, TAB+T_SB17, PS, 128,128,17, 17,128,128, 2176,0,HW, NTOK, 1.f,1.f,0);
        hipLaunchKernelGGL(proj_skip_kernel, dim3(gblocks(NHW)), dim3(256), 0, stream,
                           PS, ATTP, proj_skw, proj_skb, x, U0);
        hipLaunchKernelGGL(instnorm_kernel, dim3(NTOK), dim3(256), 0, stream,
                           U0, (const float*)nullptr, U1, ang, anb);   // attn residual out
    }

    // 7) mixer: m0 = IN(attn), then two shared 1->1 FNO layers
    hipLaunchKernelGGL(instnorm_kernel, dim3(NTOK), dim3(256), 0, stream,
                       U1, (const float*)nullptr, U2, n2g, n2b);

    auto fno_mixer = [&](const float* in, const float* w, const float* skw, const float* skb,
                         const float* ng, const float* nb, float* outb, int act) {
        float* Yre = SCRA, *Yim = SCRA + (size_t)NTOK * 2176;
        gemm(stream, in, TAB+T_CW33, Yre, 128,17,128, 128,33,17, HW,0,2176, NTOK, 1.f,0.f,0);
        gemm(stream, in, TAB+T_SW33, Yim, 128,17,128, 128,33,17, HW,0,2176, NTOK, 1.f,0.f,0);
        float* Fre = G, *Fim = G + (size_t)NTOK * 544;
        gemm(stream, TAB+T_CH32, Yre, Fre, 32,17,128, 128,17,17, 0,2176,544, NTOK,  1.f,0.f,0);
        gemm(stream, TAB+T_SH32, Yim, Fre, 32,17,128, 128,17,17, 0,2176,544, NTOK,  1.f,1.f,0);
        gemm(stream, TAB+T_CH32, Yim, Fim, 32,17,128, 128,17,17, 0,2176,544, NTOK,  1.f,0.f,0);
        gemm(stream, TAB+T_SH32, Yre, Fim, 32,17,128, 128,17,17, 0,2176,544, NTOK, -1.f,1.f,0);
        float* Mre = MK, *Mim = MK + (size_t)NTOK * 544;
        hipLaunchKernelGGL(mixer_modes_kernel, dim3(gblocks((long long)NTOK*544)), dim3(256), 0, stream,
                           Fre, Fim, w, Mre, Mim);
        float* Gre = SCRA, *Gim = SCRA + (size_t)NTOK * 2176;
        gemm(stream, TAB+T_CI32, Mre, Gre, 128,17,32, 32,17,17, 0,544,2176, NTOK,  1.f,0.f,0);
        gemm(stream, TAB+T_SI32, Mim, Gre, 128,17,32, 32,17,17, 0,544,2176, NTOK, -1.f,1.f,0);
        gemm(stream, TAB+T_SI32, Mre, Gim, 128,17,32, 32,17,17, 0,544,2176, NTOK,  1.f,0.f,0);
        gemm(stream, TAB+T_CI32, Mim, Gim, 128,17,32, 32,17,17, 0,544,2176, NTOK,  1.f,1.f,0);
        gemm(stream, Gre, TAB+T_CB17, PS, 128,128,17, 17,128,128, 2176,0,HW, NTOK, 1.f,0.f,0);
        gemm(stream, Gim, TAB+T_SB17, PS, 128,128,17, 17,128,128, 2176,0,HW, NTOK, 1.f,1.f,0);
        hipLaunchKernelGGL(instnorm_kernel, dim3(NTOK), dim3(256), 0, stream,
                           PS, (const float*)nullptr, U3, ng, nb);
        hipLaunchKernelGGL(skip_add_act_kernel, dim3(gblocks(NHW)), dim3(256), 0, stream,
                           U3, in, skw, skb, outb, NHW, act);
    };
    fno_mixer(U2, mw1, msw1, msb1, mng1, mnb1, U4, 1);   // gelu
    fno_mixer(U4, mw2, msw2, msb2, mng2, mnb2, U5, 0);

    // 8) out = IN(m2) + attn_residual
    hipLaunchKernelGGL(instnorm_kernel, dim3(NTOK), dim3(256), 0, stream,
                       U5, (const float*)nullptr, U0, mog, mob);
    hipLaunchKernelGGL(add_kernel, dim3(gblocks(NHW)), dim3(256), 0, stream,
                       U0, U1, outp, NHW);
}
